// MeshGNN_59897613910460
// MI455X (gfx1250) — matrix operand order repaired and checked by reading, not compiled, not run on hardware
//
#include <hip/hip_runtime.h>
#include <hip/hip_bf16.h>

typedef __attribute__((ext_vector_type(16))) _Float16 v16h;
typedef __attribute__((ext_vector_type(8)))  float    v8f;

// ---------------------------------------------------------------------------
// WMMA fragment layouts (CDNA5 ISA 7.12.2, wave32), f16 16x16x32:
//  A: lane L = row M=L&15; elems 0..7 -> K = (L>>4)*8 + i,
//                          elems 8..15 -> K = (L>>4)*8 + 16 + (i-8).
//  B: lane L = col N=L&15; elems 0..15 -> K in the same order as A.
//  C/D f32: elem r of lane L is (M = r + 8*(L>>4), N = L&15).
//
// LDS holds fragments PRE-SWIZZLED: frag f, lane l, elem e at
//   halfword index  (f*32 + l)*16 + e
// so a fragment load is one 32-byte contiguous read (2 x ds_load_b128).
// ---------------------------------------------------------------------------

// Stage a KxNCOLS row-major f32 weight matrix into B-fragment-swizzled f16 LDS.
// K1 = K padded to multiple of 32 (pad rows are zero).
template <int K1, int KIN, int NCOLS>
__device__ __forceinline__ void stage_b_swizzled(_Float16* __restrict__ dst,
                                                 const float* __restrict__ W,
                                                 int tid) {
  for (int i = tid; i < K1 * NCOLS; i += 128) {
    const int e  = i & 15;
    const int l  = (i >> 4) & 31;
    const int f  = i >> 9;                 // frag = kc*(NCOLS/16) + nt
    const int nt = f % (NCOLS / 16);
    const int kc = f / (NCOLS / 16);
    const int n  = nt * 16 + (l & 15);
    const int k  = kc * 32 + ((e < 8) ? ((l >> 4) * 8 + e) : ((l >> 4) * 8 + 8 + e));
    dst[i] = (_Float16)((k < KIN) ? W[k * NCOLS + n] : 0.0f);
  }
}

// For an activation value at (row M in 0..15, feature k), halfword index in
// A-fragment-swizzled storage.
__device__ __forceinline__ int a_swizzle_index(int M, int k) {
  const int kc = k >> 5;
  const int c  = k & 31;
  const int h  = (c >> 3) & 1;                       // lane half
  const int e  = (c < 16) ? (c - h * 8) : (8 + (c - 16) - h * 8);
  const int l  = (h << 4) + M;
  return (kc * 32 + l) * 16 + e;
}

__device__ __forceinline__ v16h load_frag(const _Float16* base, int frag,
                                          int lane) {
  return *(const v16h*)(base + ((frag << 5) + lane) * 16);
}

// ---------------------------------------------------------------------------
// Fused  out = LayerNorm( SiLU( relu(X@W1 + B1) @ W2 + B2 ) ), out dim = 64.
// KIN = real input dim, K1 = KIN padded to mult of 32, NH = hidden dim.
// 128 threads = 4 waves, one 16-row tile per wave.
// ---------------------------------------------------------------------------
template <int KIN, int K1, int NH>
__global__ __launch_bounds__(128) void mlp_silu_ln_kernel(
    const float* __restrict__ X, int n_rows,
    const float* __restrict__ W1, const float* __restrict__ B1,
    const float* __restrict__ W2, const float* __restrict__ B2,
    float* __restrict__ OUT) {
  __shared__ __align__(32) _Float16 sW1[K1 * NH];    // B-swizzled
  __shared__ __align__(32) _Float16 sW2[NH * 64];    // B-swizzled
  __shared__ __align__(32) _Float16 sA[4][16 * K1];  // A-swizzled input tile
  __shared__ __align__(32) _Float16 sZ[4][16 * NH];  // A-swizzled hidden act

  const int tid  = threadIdx.x;
  const int wave = tid >> 5;
  const int lane = tid & 31;

  stage_b_swizzled<K1, KIN, NH>(sW1, W1, tid);
  stage_b_swizzled<NH, NH, 64>(sW2, W2, tid);

  // per-wave input tile, stored A-swizzled (zero padding for rows/K overflow)
  const int row0 = (blockIdx.x * 4 + wave) * 16;
  for (int i = lane; i < 16 * K1; i += 32) {
    const int e  = i & 15;
    const int l  = (i >> 4) & 31;
    const int kc = i >> 9;
    const int m  = l & 15;
    const int h  = l >> 4;
    const int k  = kc * 32 + ((e < 8) ? (h * 8 + e) : (h * 8 + 8 + e));
    const int gr = row0 + m;
    const float v =
        (gr < n_rows && k < KIN) ? X[(size_t)gr * KIN + k] : 0.0f;
    sA[wave][i] = (_Float16)v;
  }
  __syncthreads();

  const v8f zf = {};
  const int mh = (lane >> 4) << 3;  // 0 or 8: row-half of C fragment
  const int nc = lane & 15;         // column within N-tile
  constexpr int NT1 = NH / 16;

  // ---- GEMM1: (16 x K1) @ (K1 x NH), + bias, ReLU -> sZ (A-swizzled) ----
  v8f acc1[NT1];
#pragma unroll
  for (int nt = 0; nt < NT1; ++nt) acc1[nt] = zf;
#pragma unroll
  for (int kc = 0; kc < K1 / 32; ++kc) {
    const v16h a = load_frag(sA[wave], kc, lane);
#pragma unroll
    for (int nt = 0; nt < NT1; ++nt) {
      const v16h b = load_frag(sW1, kc * NT1 + nt, lane);
      acc1[nt] = __builtin_amdgcn_wmma_f32_16x16x32_f16(
          false, a, false, b, (short)0, acc1[nt], false, false);
    }
  }
#pragma unroll
  for (int nt = 0; nt < NT1; ++nt) {
    const float bias = B1[nt * 16 + nc];
#pragma unroll
    for (int r = 0; r < 8; ++r) {
      float v = fmaxf(acc1[nt][r] + bias, 0.0f);
      sZ[wave][a_swizzle_index(r + mh, nt * 16 + nc)] = (_Float16)v;
    }
  }
  __syncthreads();

  // ---- GEMM2: (16 x NH) @ (NH x 64) ----
  v8f acc2[4];
#pragma unroll
  for (int nt = 0; nt < 4; ++nt) acc2[nt] = zf;
#pragma unroll
  for (int kc = 0; kc < NH / 32; ++kc) {
    const v16h a = load_frag(sZ[wave], kc, lane);
#pragma unroll
    for (int nt = 0; nt < 4; ++nt) {
      const v16h b = load_frag(sW2, kc * 4 + nt, lane);
      acc2[nt] = __builtin_amdgcn_wmma_f32_16x16x32_f16(
          false, a, false, b, (short)0, acc2[nt], false, false);
    }
  }

  // ---- bias + SiLU ----
  float vals[4][8];
#pragma unroll
  for (int nt = 0; nt < 4; ++nt) {
    const float bias = B2[nt * 16 + nc];
#pragma unroll
    for (int r = 0; r < 8; ++r) {
      float v = acc2[nt][r] + bias;
      v = v / (1.0f + __expf(-v));  // x * sigmoid(x)
      vals[nt][r] = v;
    }
  }

  // ---- LayerNorm over the 64 features of each row ----
  // Row M = r + mh lives entirely in one 16-lane half; xor masks 1..8 stay
  // inside that half.
#pragma unroll
  for (int r = 0; r < 8; ++r) {
    float s = 0.0f, q = 0.0f;
#pragma unroll
    for (int nt = 0; nt < 4; ++nt) {
      s += vals[nt][r];
      q += vals[nt][r] * vals[nt][r];
    }
#pragma unroll
    for (int msk = 8; msk >= 1; msk >>= 1) {
      s += __shfl_xor(s, msk, 32);
      q += __shfl_xor(q, msk, 32);
    }
    const float mean = s * (1.0f / 64.0f);
    const float var  = q * (1.0f / 64.0f) - mean * mean;
    const float rstd = rsqrtf(var + 1e-5f);
    const int gr = row0 + r + mh;
    if (gr < n_rows) {
#pragma unroll
      for (int nt = 0; nt < 4; ++nt)
        OUT[(size_t)gr * 64 + nt * 16 + nc] = (vals[nt][r] - mean) * rstd;
    }
  }
}

// ---------------------------------------------------------------------------
// Elementwise / graph kernels
// ---------------------------------------------------------------------------
__global__ void zero_kernel(float4* __restrict__ p, int n4) {
  const int i = blockIdx.x * blockDim.x + threadIdx.x;
  if (i < n4) p[i] = make_float4(0.f, 0.f, 0.f, 0.f);
}

__global__ void count_kernel(const int* __restrict__ dst,
                             float* __restrict__ cnt, int E) {
  const int e = blockIdx.x * blockDim.x + threadIdx.x;
  if (e < E) unsafeAtomicAdd(&cnt[dst[e]], 1.0f);
}

// thread -> (edge, 4-feature chunk): 16 consecutive lanes cover one 256B row.
__global__ void scatter_kernel(const float4* __restrict__ m,
                               const int* __restrict__ src,
                               const int* __restrict__ dst,
                               float* __restrict__ agg, int E) {
  const int i = blockIdx.x * blockDim.x + threadIdx.x;
  if (i >= (E << 4)) return;
  const int e = i >> 4;
  const int c = i & 15;
  const float4 v = m[(size_t)src[e] * 16 + c];
  float* a = agg + (size_t)dst[e] * 64 + c * 4;
  unsafeAtomicAdd(a + 0, v.x);
  unsafeAtomicAdd(a + 1, v.y);
  unsafeAtomicAdd(a + 2, v.z);
  unsafeAtomicAdd(a + 3, v.w);
}

__global__ void combine_kernel(float4* __restrict__ h,
                               const float4* __restrict__ agg,
                               const float* __restrict__ cnt, int n4) {
  const int i = blockIdx.x * blockDim.x + threadIdx.x;
  if (i < n4) {
    const float rd = 1.0f / fmaxf(cnt[i >> 4], 1.0f);
    float4 a = agg[i];
    float4 v = h[i];
    v.x += a.x * rd;
    v.y += a.y * rd;
    v.z += a.z * rd;
    v.w += a.w * rd;
    h[i] = v;
  }
}

// ---------------------------------------------------------------------------
extern "C" void kernel_launch(void* const* d_in, const int* in_sizes, int n_in,
                              void* d_out, int out_size, void* d_ws,
                              size_t ws_size, hipStream_t stream) {
  const float* x  = (const float*)d_in[0];
  const int*   ei = (const int*)d_in[1];
  const float* w0 = (const float*)d_in[2];
  const float* b0 = (const float*)d_in[3];
  const float* w1 = (const float*)d_in[4];
  const float* b1 = (const float*)d_in[5];
  const float* wa = (const float*)d_in[6];
  const float* ba = (const float*)d_in[7];
  const float* wb = (const float*)d_in[8];
  const float* bb = (const float*)d_in[9];

  const int N = in_sizes[0] / 16;          // 50000
  const int E = in_sizes[1] / 2;           // 800000
  const int L = in_sizes[6] / (64 * 128);  // 4
  const int* src = ei;
  const int* dst = ei + E;

  float* h   = (float*)d_out;              // N x 64 (running state)
  float* m   = (float*)d_ws;               // N x 64
  float* agg = m + (size_t)N * 64;         // N x 64
  float* cnt = agg + (size_t)N * 64;       // N

  // degree counts (recomputed each call; ws contents are rewritten)
  zero_kernel<<<(N / 4 + 255) / 256, 256, 0, stream>>>((float4*)cnt, N / 4);
  count_kernel<<<(E + 255) / 256, 256, 0, stream>>>(dst, cnt, E);

  // initial embedding: 16 -> 64 -> 64
  mlp_silu_ln_kernel<16, 32, 64>
      <<<(N + 63) / 64, 128, 0, stream>>>(x, N, w0, b0, w1, b1, h);

  const int nf4 = N * 16;                  // N*64 floats as float4
  for (int l = 0; l < L; ++l) {
    mlp_silu_ln_kernel<64, 64, 128><<<(N + 63) / 64, 128, 0, stream>>>(
        h, N, wa + (size_t)l * 64 * 128, ba + (size_t)l * 128,
        wb + (size_t)l * 128 * 64, bb + (size_t)l * 64, m);
    zero_kernel<<<(nf4 + 255) / 256, 256, 0, stream>>>((float4*)agg, nf4);
    scatter_kernel<<<(E * 16 + 255) / 256, 256, 0, stream>>>(
        (const float4*)m, src, dst, agg, E);
    combine_kernel<<<(nf4 + 255) / 256, 256, 0, stream>>>(
        (float4*)h, (const float4*)agg, cnt, nf4);
  }
}
